// BCS_68882685493819
// MI455X (gfx1250) — matrix operand order repaired and checked by reading, not compiled, not run on hardware
//
#include <hip/hip_runtime.h>
#include <hip/hip_bf16.h>

// ---------------------------------------------------------------------------
// GAT (4 layers) + MLP head for MI455X / gfx1250.
// GEMMs use V_WMMA_F32_16X16X32_F16 (wave32, f32 accumulate). Each wave owns a
// 32x32 output tile (2x2 WMMA fragments) so every K-step issues 8 b128 loads
// for 4 WMMAs (2 loads/WMMA, vs 4 with a 16x16 tile). f16 operands are staged
// in padded row-major (X[Nn,Kp]) / transposed (Wt[Cp,Kp]) buffers so each
// lane's fragment is contiguous 16B chunks (k = 16*hi + s, same order in A and B).
// ---------------------------------------------------------------------------

#define Nn   2048
#define Ne   67584
#define INND 128
#define HIDD 89
#define NHEAD 8
#define GOUTD 128
#define FIN  (Nn * GOUTD + INND)   // 262272
#define XLEN (Nn * GOUTD)          // 262144

typedef __attribute__((ext_vector_type(16))) _Float16 v16h;
typedef __attribute__((ext_vector_type(8)))  float    v8f;

// ---------------- helpers ----------------

__device__ __forceinline__ float atomicMaxF32(float* addr, float val) {
    int* ai = (int*)addr;
    int old = __float_as_int(*addr);
    while (__int_as_float(old) < val) {
        int assumed = old;
        old = atomicCAS(ai, assumed, __float_as_int(val));
        if (old == assumed) break;
    }
    return __int_as_float(old);
}

__device__ __forceinline__ v16h ld_frag_a(const _Float16* p) {
    // A fragment: elements 0..15 = p[0..15] (contiguous)
    v16h a;
    reinterpret_cast<float4*>(&a)[0] = *reinterpret_cast<const float4*>(p);
    reinterpret_cast<float4*>(&a)[1] = *reinterpret_cast<const float4*>(p + 8);
    return a;
}

__device__ __forceinline__ v16h ld_frag_b(const _Float16* p) {
    // B fragment: elements 0..15 = p[0..15] (contiguous)
    v16h b;
    reinterpret_cast<float4*>(&b)[0] = *reinterpret_cast<const float4*>(p);
    reinterpret_cast<float4*>(&b)[1] = *reinterpret_cast<const float4*>(p + 8);
    return b;
}

// ---------------- conversion kernels ----------------

// in: [Nn, K] f32  ->  X: [Nn, Kp] f16 (zero padded), optional ELU on input
__global__ void cvt_x_f16(const float* __restrict__ in, _Float16* __restrict__ X,
                          int K, int Kp, int elu) {
    int t = blockIdx.x * blockDim.x + threadIdx.x;
    if (t >= Nn * Kp) return;
    int n = t / Kp, k = t % Kp;
    float v = 0.f;
    if (k < K) {
        v = in[(size_t)n * K + k];
        if (elu) v = (v > 0.f) ? v : (__expf(v) - 1.f);
    }
    X[t] = (_Float16)v;
}

// W: [K, C] f32 row-major -> Wt: [Cp, Kp] f16 (transposed, zero padded)
__global__ void cvt_w_f16(const float* __restrict__ W, _Float16* __restrict__ Wt,
                          int K, int C, int Kp, int Cp) {
    int t = blockIdx.x * blockDim.x + threadIdx.x;
    if (t >= Cp * Kp) return;
    int c = t / Kp, k = t % Kp;
    Wt[t] = (_Float16)((k < K && c < C) ? W[(size_t)k * C + c] : 0.f);
}

// ------------- WMMA GEMM: out[Nn,C] = X[Nn,Kp] * Wt[Cp,Kp]^T (32x32/wave) ----
__global__ void gat_gemm_wmma(const _Float16* __restrict__ X,
                              const _Float16* __restrict__ Wt,
                              float* __restrict__ out,
                              int Kp, int C, int Cp, int ntiles) {
    int wid  = threadIdx.x >> 5;
    int tile = blockIdx.x * (blockDim.x >> 5) + wid;
    if (tile >= ntiles) return;                 // wave-uniform: EXEC stays all-1s
    int ntn = Cp >> 5;                          // 32-wide column supertiles
    int tm = tile / ntn, tn = tile % ntn;
    int lane = threadIdx.x & 31;
    int hi = lane >> 4, lo = lane & 15;

    const _Float16* xr0 = X  + (size_t)(tm * 32 + lo) * Kp;      // A rows M=lo
    const _Float16* xr1 = xr0 + (size_t)16 * Kp;                 // A rows M=16+lo
    const _Float16* wr0 = Wt + (size_t)(tn * 32 + lo) * Kp;      // B cols N=lo
    const _Float16* wr1 = wr0 + (size_t)16 * Kp;                 // B cols N=16+lo

    v8f acc00 = {}, acc01 = {}, acc10 = {}, acc11 = {};
    int ao = hi * 16, bo = hi * 16;
    for (int k0 = 0; k0 < Kp; k0 += 32) {
        const _Float16* ap0 = xr0 + k0 + ao;
        const _Float16* ap1 = xr1 + k0 + ao;
        const _Float16* bp0 = wr0 + k0 + bo;
        const _Float16* bp1 = wr1 + k0 + bo;
        v16h a0 = ld_frag_a(ap0);
        v16h a1 = ld_frag_a(ap1);
        v16h b0 = ld_frag_b(bp0);
        v16h b1 = ld_frag_b(bp1);
        __builtin_prefetch(ap0 + 32, 0, 3);      // global_prefetch_b8, near-cache
        __builtin_prefetch(ap1 + 32, 0, 3);
        __builtin_prefetch(bp0 + 32, 0, 3);
        __builtin_prefetch(bp1 + 32, 0, 3);
        acc00 = __builtin_amdgcn_wmma_f32_16x16x32_f16(false, a0, false, b0, (short)0, acc00, false, false);
        acc01 = __builtin_amdgcn_wmma_f32_16x16x32_f16(false, a0, false, b1, (short)0, acc01, false, false);
        acc10 = __builtin_amdgcn_wmma_f32_16x16x32_f16(false, a1, false, b0, (short)0, acc10, false, false);
        acc11 = __builtin_amdgcn_wmma_f32_16x16x32_f16(false, a1, false, b1, (short)0, acc11, false, false);
    }
    // D layout: VGPR r -> row (r + 8*hi), col = lane lo
    int c0 = tn * 32 + lo, c1 = c0 + 16;
    int r0 = tm * 32 + 8 * hi, r1 = r0 + 16;
    if (c0 < C) {
        #pragma unroll
        for (int r = 0; r < 8; ++r) out[(size_t)(r0 + r) * C + c0] = acc00[r];
        #pragma unroll
        for (int r = 0; r < 8; ++r) out[(size_t)(r1 + r) * C + c0] = acc10[r];
    }
    if (c1 < C) {
        #pragma unroll
        for (int r = 0; r < 8; ++r) out[(size_t)(r0 + r) * C + c1] = acc01[r];
        #pragma unroll
        for (int r = 0; r < 8; ++r) out[(size_t)(r1 + r) * C + c1] = acc11[r];
    }
}

// ---------------- attention ----------------

// el[n,h] = sum_d ft[n, h*d+dd]*al[h,dd]; er likewise
__global__ void attn_scores(const float* __restrict__ ft,
                            const float* __restrict__ al, const float* __restrict__ ar,
                            float* __restrict__ el, float* __restrict__ er,
                            int heads, int d, int C) {
    int t = blockIdx.x * blockDim.x + threadIdx.x;
    if (t >= Nn * heads) return;
    int n = t / heads, h = t % heads;
    const float* fr = ft + (size_t)n * C + h * d;
    const float* av = al + h * d;
    const float* bv = ar + h * d;
    float sa = 0.f, sb = 0.f;
    for (int dd = 0; dd < d; ++dd) { float v = fr[dd]; sa += v * av[dd]; sb += v * bv[dd]; }
    el[t] = sa; er[t] = sb;
}

// out[n,:] = bias; m = -inf; s = 0
__global__ void init_layer(float* __restrict__ outp, const float* __restrict__ b,
                           float* __restrict__ m, float* __restrict__ s,
                           int C, int heads) {
    int t = blockIdx.x * blockDim.x + threadIdx.x;
    if (t >= Nn * C) return;
    outp[t] = b[t % C];
    if (t < Nn * heads) { m[t] = -1e30f; s[t] = 0.f; }
}

__global__ void edge_max(const float* __restrict__ el, const float* __restrict__ er,
                         const int* __restrict__ src, const int* __restrict__ dst,
                         float* __restrict__ m, int heads) {
    int t = blockIdx.x * blockDim.x + threadIdx.x;
    if (t >= Ne * heads) return;
    int e = t / heads, h = t % heads;
    float v = el[src[e] * heads + h] + er[dst[e] * heads + h];
    v = (v > 0.f) ? v : 0.2f * v;                     // leaky_relu(0.2)
    atomicMaxF32(&m[dst[e] * heads + h], v);
}

__global__ void edge_exp(const float* __restrict__ el, const float* __restrict__ er,
                         const int* __restrict__ src, const int* __restrict__ dst,
                         const float* __restrict__ m, float* __restrict__ s,
                         float* __restrict__ exb, int heads) {
    int t = blockIdx.x * blockDim.x + threadIdx.x;
    if (t >= Ne * heads) return;
    int e = t / heads, h = t % heads;
    int di = dst[e] * heads + h;
    float v = el[src[e] * heads + h] + er[di];
    v = (v > 0.f) ? v : 0.2f * v;
    float ex = __expf(v - m[di]);
    exb[t] = ex;
    atomicAdd(&s[di], ex);
}

__global__ void edge_msg(const float* __restrict__ ft, const float* __restrict__ exb,
                         const float* __restrict__ s,
                         const int* __restrict__ src, const int* __restrict__ dst,
                         float* __restrict__ outp, int heads, int d, int C) {
    int t = blockIdx.x * blockDim.x + threadIdx.x;
    if (t >= Ne * heads) return;
    int e = t / heads, h = t % heads;
    int sN = src[e], dN = dst[e];
    float alpha = exb[t] / s[dN * heads + h];
    const float* fr = ft   + (size_t)sN * C + h * d;
    float*       od = outp + (size_t)dN * C + h * d;
    for (int dd = 0; dd < d; ++dd) atomicAdd(&od[dd], fr[dd] * alpha);
}

// ---------------- MLP head ----------------

__global__ void zero_f32(float* __restrict__ p, int n) {
    int t = blockIdx.x * blockDim.x + threadIdx.x;
    if (t < n) p[t] = 0.f;
}

// z1[j] += sum_{i in chunk} z[i]*w[i,500+j]; coalesced weight reads (j along threads)
__global__ void fc1_acc(const float* __restrict__ x3, const float* __restrict__ f,
                        const float* __restrict__ w, float* __restrict__ z1) {
    int tid = threadIdx.x;                   // 256 threads
    int rows = (FIN + gridDim.x - 1) / gridDim.x;
    int i0 = blockIdx.x * rows;
    int i1 = i0 + rows; if (i1 > FIN) i1 = FIN;
    float a0 = 0.f, a1 = 0.f;
    for (int i = i0; i < i1; ++i) {
        float zv = (i < XLEN) ? x3[i] : f[i - XLEN];
        const float* wr = w + (size_t)i * 500;
        a0 += zv * wr[tid];
        if (tid + 256 < 500) a1 += zv * wr[tid + 256];
    }
    atomicAdd(&z1[tid], a0);
    if (tid + 256 < 500) atomicAdd(&z1[tid + 256], a1);
}

__global__ void fc_tail(const float* __restrict__ z1, const float* __restrict__ b1,
                        const float* __restrict__ w2, const float* __restrict__ b2,
                        const float* __restrict__ w3, const float* __restrict__ b3,
                        float* __restrict__ out) {
    __shared__ float z1s[500];
    __shared__ float z2s[500];
    int j = threadIdx.x;                     // 512 threads
    if (j < 500) z1s[j] = fmaxf(z1[j] + b1[j], 0.f);
    __syncthreads();
    if (j < 500) {
        float a = 0.f;
        for (int i = 0; i < 500; ++i) a += z1s[i] * w2[(size_t)i * 500 + j];
        z2s[j] = fmaxf(a + b2[j], 0.f);
    }
    __syncthreads();
    if (j == 0) {
        float a = 0.f;
        for (int i = 0; i < 500; ++i) a += z2s[i] * w3[i];
        out[0] = a + b3[0];
    }
}

// ---------------- host orchestration ----------------

static inline int ceil_div(int a, int b) { return (a + b - 1) / b; }

extern "C" void kernel_launch(void* const* d_in, const int* in_sizes, int n_in,
                              void* d_out, int out_size, void* d_ws, size_t ws_size,
                              hipStream_t stream) {
    (void)in_sizes; (void)n_in; (void)out_size; (void)ws_size;

    const float* h    = (const float*)d_in[0];
    const float* fvec = (const float*)d_in[1];
    const int*   src  = (const int*)d_in[2];
    const int*   dst  = (const int*)d_in[3];
    const float* W[4]  = { (const float*)d_in[4],  (const float*)d_in[8],
                           (const float*)d_in[12], (const float*)d_in[16] };
    const float* al[4] = { (const float*)d_in[5],  (const float*)d_in[9],
                           (const float*)d_in[13], (const float*)d_in[17] };
    const float* ar[4] = { (const float*)d_in[6],  (const float*)d_in[10],
                           (const float*)d_in[14], (const float*)d_in[18] };
    const float* bb[4] = { (const float*)d_in[7],  (const float*)d_in[11],
                           (const float*)d_in[15], (const float*)d_in[19] };
    const float* fc1_w = (const float*)d_in[20];
    const float* fc1_b = (const float*)d_in[21];
    const float* fc2_w = (const float*)d_in[22];
    const float* fc2_b = (const float*)d_in[23];
    const float* fc3_w = (const float*)d_in[24];
    const float* fc3_b = (const float*)d_in[25];

    // ---- workspace layout (256B aligned slabs) ----
    char* ws = (char*)d_ws;
    size_t off = 0;
    auto slab = [&](size_t bytes) -> char* {
        char* p = ws + off;
        off = (off + bytes + 255) & ~(size_t)255;
        return p;
    };
    _Float16* Xf16 = (_Float16*)slab((size_t)Nn * 736 * sizeof(_Float16));
    _Float16* Wt   = (_Float16*)slab((size_t)736 * 736 * sizeof(_Float16));
    float* ft   = (float*)slab((size_t)Nn * 712 * sizeof(float));
    float* outb = (float*)slab((size_t)Nn * 712 * sizeof(float));
    float* el   = (float*)slab((size_t)Nn * NHEAD * sizeof(float));
    float* er   = (float*)slab((size_t)Nn * NHEAD * sizeof(float));
    float* mbuf = (float*)slab((size_t)Nn * NHEAD * sizeof(float));
    float* sbuf = (float*)slab((size_t)Nn * NHEAD * sizeof(float));
    float* exb  = (float*)slab((size_t)Ne * NHEAD * sizeof(float));
    float* x3   = (float*)slab((size_t)Nn * GOUTD * sizeof(float));
    float* z1   = (float*)slab(512 * sizeof(float));

    struct Cfg { int K, Kp, C, Cp, heads, d, elu_in; const float* in; float* out; };
    Cfg cfg[4] = {
        { INND,         128, HIDD * NHEAD, 736, NHEAD, HIDD,  0, h,    outb },
        { HIDD * NHEAD, 736, HIDD * NHEAD, 736, NHEAD, HIDD,  1, outb, outb },
        { HIDD * NHEAD, 736, HIDD * NHEAD, 736, NHEAD, HIDD,  1, outb, outb },
        { HIDD * NHEAD, 736, GOUTD,        128, 1,     GOUTD, 1, outb, x3  },
    };

    const int TB = 256;
    for (int l = 0; l < 4; ++l) {
        const Cfg& c = cfg[l];
        // 1) activations -> f16 (ELU fused for layers 1..3 inputs)
        cvt_x_f16<<<ceil_div(Nn * c.Kp, TB), TB, 0, stream>>>(c.in, Xf16, c.K, c.Kp, c.elu_in);
        // 2) weights -> f16 transposed
        cvt_w_f16<<<ceil_div(c.Cp * c.Kp, TB), TB, 0, stream>>>(W[l], Wt, c.K, c.C, c.Kp, c.Cp);
        // 3) WMMA GEMM: ft = X * W   (32x32 tile per wave)
        int ntiles = (Nn / 32) * (c.Cp / 32);
        gat_gemm_wmma<<<ceil_div(ntiles, TB / 32), TB, 0, stream>>>(Xf16, Wt, ft, c.Kp, c.C, c.Cp, ntiles);
        // 4) attention scores
        attn_scores<<<ceil_div(Nn * c.heads, TB), TB, 0, stream>>>(ft, al[l], ar[l], el, er, c.heads, c.d, c.C);
        // 5) init out=bias, m=-inf, s=0
        init_layer<<<ceil_div(Nn * c.C, TB), TB, 0, stream>>>(c.out, bb[l], mbuf, sbuf, c.C, c.heads);
        // 6-8) edge softmax + message scatter
        int et = Ne * c.heads;
        edge_max<<<ceil_div(et, TB), TB, 0, stream>>>(el, er, src, dst, mbuf, c.heads);
        edge_exp<<<ceil_div(et, TB), TB, 0, stream>>>(el, er, src, dst, mbuf, sbuf, exb, c.heads);
        edge_msg<<<ceil_div(et, TB), TB, 0, stream>>>(ft, exb, sbuf, src, dst, c.out, c.heads, c.d, c.C);
    }

    // MLP head
    zero_f32<<<2, 256, 0, stream>>>(z1, 512);
    fc1_acc<<<512, 256, 0, stream>>>(x3, fvec, fc1_w, z1);
    fc_tail<<<1, 512, 0, stream>>>(z1, fc1_b, fc2_w, fc2_b, fc3_w, fc3_b, (float*)d_out);
}
